// CodebookModule_75342316306558
// MI455X (gfx1250) — compile-verified, matrix operand order checked by reading, not run on hardware
//
#include <hip/hip_runtime.h>
#include <hip/hip_bf16.h>
#include <stdint.h>

// ---------------------------------------------------------------------------
// Cosine-sim VQ + gumbel softmax for MI455X (gfx1250, wave32, WMMA).
//   sim GEMM: bf16 inputs (normalized unit vectors), f32 accumulate via
//   v_wmma_f32_16x16x32_bf16. Memory-bound overall (~258MB HBM @ 23.3TB/s).
// ---------------------------------------------------------------------------

typedef __attribute__((ext_vector_type(16))) __bf16 v16bf;
typedef __attribute__((ext_vector_type(8)))  __bf16 v8bf;
typedef __attribute__((ext_vector_type(8)))  float  v8f;

#define DDIM 512
#define KDIM 1024
#define EPSN 1e-8f

__device__ __forceinline__ float wave_sum(float v) {
#pragma unroll
  for (int m = 16; m >= 1; m >>= 1) v += __shfl_xor(v, m, 32);
  return v;
}
__device__ __forceinline__ float wave_max(float v) {
#pragma unroll
  for (int m = 16; m >= 1; m >>= 1) v = fmaxf(v, __shfl_xor(v, m, 32));
  return v;
}

// splitmix64-based counter RNG -> uniform in (0,1) -> gumbel
__device__ __forceinline__ float gumbel_noise(uint64_t ctr, uint64_t seed) {
  uint64_t z = ctr * 0x9E3779B97F4A7C15ULL + seed;
  z = (z ^ (z >> 30)) * 0xBF58476D1CE4E5B9ULL;
  z = (z ^ (z >> 27)) * 0x94D049BB133111EBULL;
  z ^= z >> 31;
  uint32_t m24 = (uint32_t)(z >> 40);                      // 24 random bits
  float u = ((float)m24 + 0.5f) * (1.0f / 16777216.0f);    // (0,1)
  return -__logf(-__logf(u));
}

// ---------------------------------------------------------------------------
// Kernel 1: normalize codebook rows, store bf16 copy (1 MB, L2-resident).
// 8 waves per block, one row per wave, wave32 shfl reduction.
// ---------------------------------------------------------------------------
__global__ void __launch_bounds__(256)
vq_normalize_codebook(const float* __restrict__ cb, __bf16* __restrict__ cn) {
  const int wave = threadIdx.x >> 5;
  const int lane = threadIdx.x & 31;
  const int row  = blockIdx.x * 8 + wave;
  const float* src = cb + (size_t)row * DDIM;
  float v[16];
  float ss = 0.f;
#pragma unroll
  for (int i = 0; i < 16; ++i) {
    v[i] = src[lane + 32 * i];
    ss += v[i] * v[i];
  }
  ss = wave_sum(ss);
  const float rinv = 1.0f / fmaxf(sqrtf(ss), EPSN);
  __bf16* dst = cn + (size_t)row * DDIM;
#pragma unroll
  for (int i = 0; i < 16; ++i) dst[lane + 32 * i] = (__bf16)(v[i] * rinv);
}

// ---------------------------------------------------------------------------
// Kernel 2: per block: 16 rows. normalize x -> LDS bf16, WMMA sim tile
// (16x1024), gumbel argmax + softmax, z_q gather, index.
// ---------------------------------------------------------------------------
__global__ void __launch_bounds__(256)
vq_main(const float* __restrict__ x, const float* __restrict__ cb,
        const __bf16* __restrict__ cn, float* __restrict__ out_zq,
        float* __restrict__ out_w, float* __restrict__ out_idx) {
  extern __shared__ char smem[];
  float (*lds_sim)[KDIM] = (float (*)[KDIM])smem;                    // 64 KB
  __bf16 (*lds_a)[DDIM] = (__bf16 (*)[DDIM])(smem + 16 * KDIM * 4);  // 16 KB

  const int wave = threadIdx.x >> 5;
  const int lane = threadIdx.x & 31;
  const int l15  = lane & 15;
  const int hi   = lane >> 4;  // half-wave select
  const int rowBase = blockIdx.x * 16;

  // ---- Stage 1: normalize 2 x-rows per wave into LDS as bf16 ----
#pragma unroll
  for (int rr = 0; rr < 2; ++rr) {
    const int r = wave * 2 + rr;
    const float* src = x + (size_t)(rowBase + r) * DDIM;
    float v[16];
    float ss = 0.f;
#pragma unroll
    for (int i = 0; i < 16; ++i) {
      v[i] = src[lane + 32 * i];
      ss += v[i] * v[i];
    }
    ss = wave_sum(ss);
    const float rinv = 1.0f / fmaxf(sqrtf(ss), EPSN);
#pragma unroll
    for (int i = 0; i < 16; ++i) lds_a[r][lane + 32 * i] = (__bf16)(v[i] * rinv);
  }
  __syncthreads();

  // ---- Stage 2: 16x128 sim strip per wave via v_wmma_f32_16x16x32_bf16 ----
  v8f acc[8];
#pragma unroll
  for (int t = 0; t < 8; ++t) acc[t] = (v8f){0.f, 0.f, 0.f, 0.f, 0.f, 0.f, 0.f, 0.f};

  const int colWave = wave * 128;
  for (int kt = 0; kt < DDIM / 32; ++kt) {
    // 16-bit fragment layout (ISA 7.12.2): lanes 0-15 hold K 0-7 / 16-23,
    // lanes 16-31 hold K 8-15 / 24-31 (packed pairs per VGPR).
    const int k1 = kt * 32 + hi * 8;
    const int k2 = kt * 32 + 16 + hi * 8;
    const __bf16* arow = &lds_a[l15][0];
    const v8bf alo = *(const v8bf*)(arow + k1);
    const v8bf ahi = *(const v8bf*)(arow + k2);
    const v16bf afrag = __builtin_shufflevector(alo, ahi, 0, 1, 2, 3, 4, 5, 6, 7,
                                                8, 9, 10, 11, 12, 13, 14, 15);
#pragma unroll
    for (int t = 0; t < 8; ++t) {
      const __bf16* brow = cn + (size_t)(colWave + t * 16 + l15) * DDIM;
      const v8bf blo = *(const v8bf*)(brow + k1);
      const v8bf bhi = *(const v8bf*)(brow + k2);
      const v16bf bfrag = __builtin_shufflevector(blo, bhi, 0, 1, 2, 3, 4, 5, 6, 7,
                                                  8, 9, 10, 11, 12, 13, 14, 15);
      acc[t] = __builtin_amdgcn_wmma_f32_16x16x32_bf16(
          /*neg_a=*/false, afrag, /*neg_b=*/false, bfrag,
          /*c_mod=*/(short)0, acc[t], /*reuse_a=*/false, /*reuse_b=*/false);
    }
  }
  // C/D layout: VGPR r -> row r (lanes 0-15) / row r+8 (lanes 16-31), col = lane&15
#pragma unroll
  for (int t = 0; t < 8; ++t) {
    const int col = colWave + t * 16 + l15;
#pragma unroll
    for (int r = 0; r < 8; ++r) lds_sim[r + hi * 8][col] = acc[t][r];
  }
  __syncthreads();

  // ---- Stage 3: gumbel hard argmax + soft softmax, 2 rows per wave ----
  const uint64_t seedH = 0x243F6A8885A308D3ULL;
  const uint64_t seedS = 0x13198A2E03707344ULL;
#pragma unroll
  for (int rr = 0; rr < 2; ++rr) {
    const int r = wave * 2 + rr;
    const uint64_t rowg = (uint64_t)(rowBase + r);

    float bv = -1e30f;
    int bi = 0;
    float ms = -1e30f;
#pragma unroll
    for (int i = 0; i < KDIM / 32; ++i) {
      const int c = lane + 32 * i;
      const float s = lds_sim[r][c];
      const uint64_t ctr = rowg * (uint64_t)KDIM + (uint64_t)c;
      const float zh = s + gumbel_noise(ctr, seedH);
      const float zs = s + gumbel_noise(ctr, seedS);
      lds_sim[r][c] = zs;  // row owned exclusively by this wave
      if (zh > bv || (zh == bv && c < bi)) { bv = zh; bi = c; }
      ms = fmaxf(ms, zs);
    }
#pragma unroll
    for (int m = 16; m >= 1; m >>= 1) {
      const float ov = __shfl_xor(bv, m, 32);
      const int oi = __shfl_xor(bi, m, 32);
      if (ov > bv || (ov == bv && oi < bi)) { bv = ov; bi = oi; }
    }
    ms = wave_max(ms);

    float e[KDIM / 32];
    float sum = 0.f;
#pragma unroll
    for (int i = 0; i < KDIM / 32; ++i) {
      e[i] = __expf(lds_sim[r][lane + 32 * i] - ms);
      sum += e[i];
    }
    sum = wave_sum(sum);
    const float rsum = 1.0f / sum;
    float* wrow = out_w + rowg * KDIM;
#pragma unroll
    for (int i = 0; i < KDIM / 32; ++i) wrow[lane + 32 * i] = e[i] * rsum;

    // z_q = codebook[idx] (original, un-normalized rows)
    const float* crow = cb + (size_t)bi * DDIM;
    float* zrow = out_zq + rowg * DDIM;
#pragma unroll
    for (int i = 0; i < DDIM / 32; ++i) zrow[lane + 32 * i] = crow[lane + 32 * i];
    if (lane == 0) out_idx[rowg] = (float)bi;
  }
}

extern "C" void kernel_launch(void* const* d_in, const int* in_sizes, int n_in,
                              void* d_out, int out_size, void* d_ws, size_t ws_size,
                              hipStream_t stream) {
  (void)n_in; (void)out_size; (void)ws_size;
  const float* x  = (const float*)d_in[0];
  const float* cb = (const float*)d_in[1];
  const int B = in_sizes[0] / DDIM;  // 32768

  __bf16* cn = (__bf16*)d_ws;  // normalized codebook, bf16, KDIM*DDIM*2 = 1 MB

  float* out = (float*)d_out;
  float* out_zq  = out;                                          // B*D
  float* out_w   = out + (size_t)B * DDIM;                       // B*K
  float* out_idx = out + (size_t)B * DDIM + (size_t)B * KDIM;    // B

  vq_normalize_codebook<<<KDIM / 8, 256, 0, stream>>>(cb, cn);

  const size_t smem = (size_t)16 * KDIM * sizeof(float) + (size_t)16 * DDIM * sizeof(__bf16);
  vq_main<<<B / 16, 256, smem, stream>>>(x, cb, cn, out_zq, out_w, out_idx);
}
